// STFT_8787503088288
// MI455X (gfx1250) — matrix operand order repaired and checked by reading, not compiled, not run on hardware
//
#include <hip/hip_runtime.h>
#include <hip/hip_bf16.h>

typedef _Float16 v16h __attribute__((ext_vector_type(16)));
typedef float    v8f  __attribute__((ext_vector_type(8)));

#define T_LEN    480000
#define N_FFT    1024
#define HOP      512
#define PAD      512
#define NFRAMES  938
#define CDIM     1026
#define KPAD     1056                        // CDIM padded to 33*32
#define SA_PITCH 1058                        // LDS pitch (odd dword stride)
#define PADDED   (T_LEN + 2*PAD)            // 481024
#define CHUNKS_PER_WG 31
#define ROWS_PER_WG   32
#define NGROUPS  31
#define XWIN     (CHUNKS_PER_WG*HOP + N_FFT) // 16896 halfs
#define SMEM_BYTES (ROWS_PER_WG * N_FFT * 4) // 128 KB

// ---------------------------------------------------------------------------
// Kernel A0: fwdT[n][c] = (f16)fwd[c][n], K zero-padded to 1056, LDS-tiled
// transpose so both global read and write are contiguous.
// ---------------------------------------------------------------------------
__global__ __launch_bounds__(256) void transpose_fwd_kernel(
    const float* __restrict__ fwd, _Float16* __restrict__ fwdT)
{
  __shared__ _Float16 sT[64][65];
  const int n0 = blockIdx.x * 64;
  const int c0 = blockIdx.y * 64;
  const int tid = threadIdx.x;
  for (int idx = tid; idx < 64 * 64; idx += 256) {
    const int ci = idx >> 6, nj = idx & 63;
    const int c = c0 + ci;
    float v = (c < CDIM) ? fwd[(size_t)c * N_FFT + n0 + nj] : 0.f;
    sT[nj][ci] = (_Float16)v;
  }
  __syncthreads();
  for (int idx = tid; idx < 64 * 64; idx += 256) {
    const int nj = idx >> 6, ci = idx & 63;
    const int c = c0 + ci;
    if (c < KPAD)
      fwdT[(size_t)(n0 + nj) * KPAD + c] = sT[nj][ci];
  }
}

// ---------------------------------------------------------------------------
// Kernel A1: GT[col][n] = sum_c inv[c][col] * fwd[c][n] via WMMA.
// Workgroup: 32 cols (2 M-tiles) x 256 n (8 waves x 2 N-tiles).
// inv col-slice staged once into LDS transposed (A side); fwdT read directly
// from global as contiguous v16h B-fragments.
// ---------------------------------------------------------------------------
__global__ __launch_bounds__(256) void build_gt_wmma_kernel(
    const float* __restrict__ invb, const _Float16* __restrict__ fwdT,
    _Float16* __restrict__ gt)
{
  extern __shared__ char smemA[];
  _Float16* sA = (_Float16*)smemA;           // [32][SA_PITCH] halfs

  const int col0 = blockIdx.x * 32;
  const int n0   = blockIdx.y * 256;
  const int tid  = threadIdx.x;
  const int lane = tid & 31;
  const int wv   = tid >> 5;

  // stage invT slice: sA[col][c], coalesced global reads (col fastest)
  for (int idx = tid; idx < 32 * KPAD; idx += 256) {
    const int c = idx >> 5, col = idx & 31;
    float v = (c < CDIM) ? invb[(size_t)c * N_FFT + col0 + col] : 0.f;
    sA[(size_t)col * SA_PITCH + c] = (_Float16)v;
  }
  __syncthreads();

  const int m    = lane & 15;
  const int hi   = lane >> 4;
  const int hi8  = hi * 8;
  const int hi16 = hi * 16;

  v8f acc[2][2];
  #pragma unroll
  for (int mt = 0; mt < 2; ++mt)
    #pragma unroll
    for (int nt = 0; nt < 2; ++nt)
      acc[mt][nt] = (v8f){0.f, 0.f, 0.f, 0.f, 0.f, 0.f, 0.f, 0.f};

  const _Float16* arow0 = sA + (size_t)m * SA_PITCH;
  const _Float16* arow1 = sA + (size_t)(m + 16) * SA_PITCH;
  const _Float16* gb    = fwdT + (size_t)(n0 + wv * 32 + m) * KPAD + hi16;

  for (int kb = 0; kb < KPAD; kb += 32) {
    v16h a0, a1;
    #pragma unroll
    for (int p = 0; p < 8; ++p) {
      const int k0 = hi16 + (p << 1);
      a0[2 * p]     = arow0[kb + k0];
      a0[2 * p + 1] = arow0[kb + k0 + 1];
      a1[2 * p]     = arow1[kb + k0];
      a1[2 * p + 1] = arow1[kb + k0 + 1];
    }
    #pragma unroll
    for (int nt = 0; nt < 2; ++nt) {
      const v16h bf = *(const v16h*)(gb + (size_t)(nt * 16) * KPAD + kb);
      acc[0][nt] = __builtin_amdgcn_wmma_f32_16x16x32_f16(
          false, a0, false, bf, (short)0, acc[0][nt], false, false);
      acc[1][nt] = __builtin_amdgcn_wmma_f32_16x16x32_f16(
          false, a1, false, bf, (short)0, acc[1][nt], false, false);
    }
  }

  // store C tiles: row M = mt*16 + r + 8*hi (=col), N = n (16 contig halfs/half-wave)
  #pragma unroll
  for (int mt = 0; mt < 2; ++mt)
    #pragma unroll
    for (int nt = 0; nt < 2; ++nt) {
      const int n = n0 + wv * 32 + nt * 16 + m;
      #pragma unroll
      for (int r = 0; r < 8; ++r) {
        const int col = col0 + mt * 16 + r + hi8;
        gt[(size_t)col * N_FFT + n] = (_Float16)acc[mt][nt][r];
      }
    }
}

// ---------------------------------------------------------------------------
// Kernel B: fused frames = X_windows @ G (f16 WMMA, f32 accum) + overlap-add
// + normalization. Workgroup g owns frame rows 31g..31g+31 and exclusively
// writes output chunks j = 31g+1..31g+31 (1/31 redundant row, no atomics).
// LDS: f16 x-window overlaid with the 32x1024 f32 frames tile -> 128 KB,
// 2 workgroups/WGP.
// ---------------------------------------------------------------------------
__global__ __launch_bounds__(256) void fused_istft_kernel(
    const float* __restrict__ x, const _Float16* __restrict__ gt,
    const float* __restrict__ wsum, float* __restrict__ out)
{
  extern __shared__ char smem[];
  _Float16* sx  = (_Float16*)smem;   // phase 1: 16896 halfs
  float*    sfr = (float*)smem;      // phase 2: 32*1024 f32 (overlaid)

  const int g    = blockIdx.x;
  const int b    = blockIdx.y;
  const int tid  = threadIdx.x;
  const int lane = tid & 31;
  const int wv   = tid >> 5;
  const long p0  = (long)g * (CHUNKS_PER_WG * HOP);

  const float* xb = x + (size_t)b * T_LEN;
  for (int i = tid; i < XWIN; i += 256) {
    const long p = p0 + i;
    float v = 0.f;
    if (p < PADDED) {
      long q;
      if      (p < PAD)         q = PAD - p;
      else if (p < PAD + T_LEN) q = p - PAD;
      else                      q = 2L * T_LEN - 2 - (p - PAD);
      v = xb[q];
    }
    sx[i] = (_Float16)v;
  }
  __syncthreads();

  const int m    = lane & 15;
  const int hi   = lane >> 4;
  const int hi8  = hi * 8;
  const int hi16 = hi * 16;

  v8f acc[2][8];
  #pragma unroll
  for (int mt = 0; mt < 2; ++mt)
    #pragma unroll
    for (int t = 0; t < 8; ++t)
      acc[mt][t] = (v8f){0.f, 0.f, 0.f, 0.f, 0.f, 0.f, 0.f, 0.f};

  const _Float16* xrow0 = sx + (size_t)m * HOP;
  const _Float16* xrow1 = sx + (size_t)(m + 16) * HOP;
  const _Float16* gbase = gt + (size_t)(wv * 128 + m) * N_FFT + hi16;

  for (int kb = 0; kb < N_FFT; kb += 32) {
    v16h a0, a1;
    #pragma unroll
    for (int p = 0; p < 8; ++p) {
      const int k0 = hi16 + (p << 1);
      a0[2 * p]     = xrow0[kb + k0];
      a0[2 * p + 1] = xrow0[kb + k0 + 1];
      a1[2 * p]     = xrow1[kb + k0];
      a1[2 * p + 1] = xrow1[kb + k0 + 1];
    }
    v16h bf[8];
    #pragma unroll
    for (int t = 0; t < 8; ++t) {
      const _Float16* bp = gbase + (size_t)(t * 16) * N_FFT + kb;
      bf[t] = *(const v16h*)bp;
      __builtin_prefetch((const void*)(bp + 32), 0, 3);
    }
    #pragma unroll
    for (int t = 0; t < 8; ++t) {
      acc[0][t] = __builtin_amdgcn_wmma_f32_16x16x32_f16(
          false, a0, false, bf[t], (short)0, acc[0][t], false, false);
      acc[1][t] = __builtin_amdgcn_wmma_f32_16x16x32_f16(
          false, a1, false, bf[t], (short)0, acc[1][t], false, false);
    }
  }

  __syncthreads();   // sx dead; sfr overlays it

  #pragma unroll
  for (int mt = 0; mt < 2; ++mt)
    #pragma unroll
    for (int t = 0; t < 8; ++t) {
      const int col = wv * 128 + t * 16 + m;
      #pragma unroll
      for (int r = 0; r < 8; ++r)
        sfr[(mt * 16 + r + hi8) * N_FFT + col] = acc[mt][t][r];
    }
  __syncthreads();

  for (int idx = tid; idx < CHUNKS_PER_WG * HOP; idx += 256) {
    const int jl = idx >> 9;
    const int i  = idx & 511;
    const int j  = g * CHUNKS_PER_WG + 1 + jl;
    if (j > NFRAMES) continue;
    float v = sfr[jl * N_FFT + HOP + i];
    if (j <= NFRAMES - 1) v += sfr[(jl + 1) * N_FFT + i];
    const int s = j * HOP + i;
    const int t = s - PAD;
    if (t < T_LEN) {
      float d = wsum[s];
      d = (d > 1.17549435e-38f) ? d : 1.0f;
      out[(size_t)b * T_LEN + t] = v * 2.0f / d;
    }
  }
}

extern "C" void kernel_launch(void* const* d_in, const int* in_sizes, int n_in,
                              void* d_out, int out_size, void* d_ws, size_t ws_size,
                              hipStream_t stream) {
  (void)in_sizes; (void)n_in; (void)out_size; (void)ws_size;
  const float* x    = (const float*)d_in[0];
  const float* fwd  = (const float*)d_in[1];   // (1026, 1024)
  const float* invb = (const float*)d_in[2];   // (1026, 1024)
  const float* wsum = (const float*)d_in[3];   // (480768,)
  float* out = (float*)d_out;

  _Float16* fwdT = (_Float16*)d_ws;                             // 1024*1056 f16
  _Float16* gt   = (_Float16*)((char*)d_ws + (size_t)N_FFT * KPAD * 2);

  transpose_fwd_kernel<<<dim3(16, 17), 256, 0, stream>>>(fwd, fwdT);

  const size_t smemA = (size_t)32 * SA_PITCH * sizeof(_Float16);
  build_gt_wmma_kernel<<<dim3(32, 4), 256, smemA, stream>>>(invb, fwdT, gt);

  fused_istft_kernel<<<dim3(NGROUPS, 32), 256, SMEM_BYTES, stream>>>(
      x, gt, wsum, out);
}
